// CompositeK_31903017074736
// MI455X (gfx1250) — matrix operand order repaired and checked by reading, not compiled, not run on hardware
//
#include <hip/hip_runtime.h>
#include <hip/hip_bf16.h>

#define BB 4
#define SS 4096
#define DD 1024
#define DE 256
#define NC 32
#define ECCN 32
#define DC 256
#define KFD 512
#define MROWS (BB*SS)

typedef __attribute__((ext_vector_type(16))) __bf16 v16bf;
typedef __attribute__((ext_vector_type(8)))  float  v8f;

union Frag32B { v16bf v; uint4 u[2]; };

__device__ __forceinline__ unsigned short f2bf(float f) {
  unsigned int u = __float_as_uint(f);
  unsigned int r = (u + 0x7FFFu + ((u >> 16) & 1u)) >> 16;   // round-to-nearest-even
  return (unsigned short)r;
}

// ---------------- staging kernels ----------------
__global__ void k_f32_to_bf16(const float* __restrict__ src, unsigned short* __restrict__ dst, int n) {
  int i = blockIdx.x * blockDim.x + threadIdx.x;
  if (i < n) dst[i] = f2bf(src[i]);
}

// W [K,N] f32 row-major -> WT [N,K] bf16 row-major
__global__ void k_transpose_bf16(const float* __restrict__ W, unsigned short* __restrict__ WT, int K, int N) {
  int i = blockIdx.x * blockDim.x + threadIdx.x;
  if (i >= K * N) return;
  int k = i / N, n = i - k * N;
  WT[(size_t)n * K + k] = f2bf(W[i]);
}

// ---- WMMA GEMM, 2x2 register-blocked: each wave computes 32x32 of C = A @ Bt^T + bias ----
__global__ void k_wmma_gemm(const unsigned short* __restrict__ A,   // [M,K] bf16
                            const unsigned short* __restrict__ Bt,  // [N,K] bf16 (W transposed)
                            const float* __restrict__ bias,         // [N] or null
                            float* __restrict__ C,                  // [M,N] f32
                            int M, int N, int K) {
  int lane = threadIdx.x & 31;
  int wave = threadIdx.x >> 5;
  int tilesM2 = M >> 5, tilesN2 = N >> 5;
  int t = blockIdx.x * 4 + wave;                  // blockDim = 128 -> 4 waves
  if (t >= tilesM2 * tilesN2) return;
  int tm = t % tilesM2, tn = t / tilesM2;         // tm-major: block's waves share B rows
  int hl = (lane < 16) ? 0 : 1;
  const unsigned short* arow0 = A  + (size_t)(tm * 32 + (lane & 15)) * K;
  const unsigned short* arow1 = arow0 + (size_t)16 * K;
  const unsigned short* brow0 = Bt + (size_t)(tn * 32 + (lane & 15)) * K;
  const unsigned short* brow1 = brow0 + (size_t)16 * K;
  v8f acc00 = {}, acc01 = {}, acc10 = {}, acc11 = {};
  for (int kk = 0; kk < K; kk += 32) {
    Frag32B a0, a1, b0, b1;
    // A 16x32 bf16: lanes0-15 K {kk..kk+7, kk+16..23}; lanes16-31 K {kk+8..15, kk+24..31}
    a0.u[0] = *(const uint4*)(arow0 + kk + hl * 8);
    a0.u[1] = *(const uint4*)(arow0 + kk + hl * 8 + 16);
    a1.u[0] = *(const uint4*)(arow1 + kk + hl * 8);
    a1.u[1] = *(const uint4*)(arow1 + kk + hl * 8 + 16);
    // B 32x16 bf16: lanes0-15 K {kk..kk+7, kk+16..23}; lanes16-31 K {kk+8..15, kk+24..31}
    b0.u[0] = *(const uint4*)(brow0 + kk + hl * 8);
    b0.u[1] = *(const uint4*)(brow0 + kk + hl * 8 + 16);
    b1.u[0] = *(const uint4*)(brow1 + kk + hl * 8);
    b1.u[1] = *(const uint4*)(brow1 + kk + hl * 8 + 16);
    acc00 = __builtin_amdgcn_wmma_f32_16x16x32_bf16(false, a0.v, false, b0.v, (short)0, acc00, false, false);
    acc01 = __builtin_amdgcn_wmma_f32_16x16x32_bf16(false, a0.v, false, b1.v, (short)0, acc01, false, false);
    acc10 = __builtin_amdgcn_wmma_f32_16x16x32_bf16(false, a1.v, false, b0.v, (short)0, acc10, false, false);
    acc11 = __builtin_amdgcn_wmma_f32_16x16x32_bf16(false, a1.v, false, b1.v, (short)0, acc11, false, false);
  }
  int col0 = tn * 32 + (lane & 15);
  int col1 = col0 + 16;
  float bv0 = bias ? bias[col0] : 0.f;
  float bv1 = bias ? bias[col1] : 0.f;
#pragma unroll
  for (int r = 0; r < 8; r++) {
    size_t row0 = (size_t)(tm * 32 + hl * 8 + r);
    size_t row1 = row0 + 16;
    C[row0 * N + col0] = acc00[r] + bv0;
    C[row0 * N + col1] = acc01[r] + bv1;
    C[row1 * N + col0] = acc10[r] + bv0;
    C[row1 * N + col1] = acc11[r] + bv1;
  }
}

// ---------------- embedding L2-normalize -> bf16 ----------------
__global__ void k_normalize(const float* __restrict__ emb, unsigned short* __restrict__ out) {
  int row = blockIdx.x, t = threadIdx.x;          // 256 threads
  __shared__ float red[256];
  float v = emb[(size_t)row * DE + t];
  red[t] = v * v;
  __syncthreads();
  for (int s = 128; s > 0; s >>= 1) {
    if (t < s) red[t] += red[t + s];
    __syncthreads();
  }
  float inv = 1.f / (sqrtf(red[0]) + 1e-8f);
  out[(size_t)row * DE + t] = f2bf(v * inv);
}

// ---------------- fused sim (WMMA) + top-32 ----------------
__global__ void k_sim_topk(const unsigned short* __restrict__ embn,   // [B*S, DE] normalized bf16
                           float* __restrict__ scores, float* __restrict__ indices,
                           float* __restrict__ minh, float* __restrict__ maxh) {
  int lane = threadIdx.x;                         // 32 threads = 1 wave per block
  int blk  = blockIdx.x;
  int b  = blk / (SS / 16);
  int mt = blk % (SS / 16);
  int m0 = mt * 16;
  int hl = (lane < 16) ? 0 : 1;

  // persistent A fragments: this wave's 16 query rows, full K=256
  const unsigned short* arow = embn + (size_t)(b * SS + m0 + (lane & 15)) * DE;
  Frag32B afr[8];
#pragma unroll
  for (int ks = 0; ks < 8; ks++) {
    int kk = ks * 32;
    afr[ks].u[0] = *(const uint4*)(arow + kk + hl * 8);
    afr[ks].u[1] = *(const uint4*)(arow + kk + hl * 8 + 16);
  }

  __shared__ float simS[16][16];
  __shared__ float topS[16][32];
  __shared__ int   topI[16][32];
  if (lane < 16)
    for (int j = 0; j < 32; j++) topS[lane][j] = -3e38f;
  float minv = -3e38f; int minp = 0;

  for (int kt = 0; kt < SS / 16; kt++) {
    const unsigned short* brow = embn + (size_t)(b * SS + kt * 16 + (lane & 15)) * DE;
    v8f acc = {};
#pragma unroll
    for (int ks = 0; ks < 8; ks++) {
      int kk = ks * 32;
      Frag32B bfr;
      bfr.u[0] = *(const uint4*)(brow + kk + hl * 8);
      bfr.u[1] = *(const uint4*)(brow + kk + hl * 8 + 16);
      acc = __builtin_amdgcn_wmma_f32_16x16x32_bf16(false, afr[ks].v, false, bfr.v, (short)0, acc, false, false);
    }
    __syncthreads();                              // previous tile's readers are done
#pragma unroll
    for (int r = 0; r < 8; r++) simS[r + hl * 8][lane & 15] = acc[r];
    __syncthreads();
    if (lane < 16) {
      int qrow = m0 + lane;
      for (int c = 0; c < 16; c++) {
        int idx = kt * 16 + c;
        if (idx == qrow) continue;                // diagonal -> -1e9, never in top-32
        float s = simS[lane][c];
        if (s > minv) {                           // rare after warm-up (~155 times / 4096)
          topS[lane][minp] = s; topI[lane][minp] = idx;
          minv = topS[lane][0]; minp = 0;
          for (int j = 1; j < 32; j++)
            if (topS[lane][j] < minv) { minv = topS[lane][j]; minp = j; }
        }
      }
    }
  }

  if (lane < 16) {
    // sort descending (insertion sort over 32)
    for (int i = 1; i < 32; i++) {
      float sv = topS[lane][i]; int iv = topI[lane][i];
      int j = i - 1;
      while (j >= 0 && topS[lane][j] < sv) {
        topS[lane][j + 1] = topS[lane][j]; topI[lane][j + 1] = topI[lane][j]; j--;
      }
      topS[lane][j + 1] = sv; topI[lane][j + 1] = iv;
    }
    size_t base = (size_t)(b * SS + m0 + lane);
    for (int j = 0; j < 32; j++) {
      scores[base * 32 + j]  = topS[lane][j];
      indices[base * 32 + j] = (float)topI[lane][j];
    }
    for (int j = 0; j < 16; j++) {
      minh[base * 16 + j] = topS[lane][j];
      maxh[base * 16 + j] = -topS[lane][16 + j];
    }
  }
}

// ---------------- build concat(embedding, coords) in bf16 ----------------
__global__ void k_build_kflat(const float* __restrict__ emb, const float* __restrict__ coords,
                              unsigned short* __restrict__ kf) {
  int i = blockIdx.x * blockDim.x + threadIdx.x;
  if (i >= MROWS * KFD) return;
  int row = i >> 9, col = i & 511;
  float v = (col < DE) ? emb[(size_t)row * DE + col] : coords[(size_t)row * DC + (col - DE)];
  kf[i] = f2bf(v);
}

// ---------------- tiny ecc MLP ----------------
__global__ void k_ecc_hidden(const float* __restrict__ p, const float* __restrict__ W1,
                             const float* __restrict__ b1, float* __restrict__ h) {
  int row = blockIdx.x, t = threadIdx.x;          // 64 threads
  __shared__ float pr[ECCN];
  if (t < ECCN) pr[t] = p[(size_t)row * ECCN + t];
  __syncthreads();
  float acc = b1[t];
#pragma unroll
  for (int k = 0; k < ECCN; k++) acc += pr[k] * W1[k * (ECCN * 2) + t];
  h[(size_t)row * (ECCN * 2) + t] = tanhf(acc);
}

__global__ void k_ecc_out(const float* __restrict__ h, const float* __restrict__ W2,
                          const float* __restrict__ b2, float* __restrict__ out) {
  int row = blockIdx.x, t = threadIdx.x;          // 32 threads
  __shared__ float hr[ECCN * 2];
  hr[t] = h[(size_t)row * (ECCN * 2) + t];
  hr[t + 32] = h[(size_t)row * (ECCN * 2) + t + 32];
  __syncthreads();
  float acc = b2[t];
#pragma unroll
  for (int k = 0; k < ECCN * 2; k++) acc += hr[k] * W2[k * ECCN + t];
  out[(size_t)row * ECCN + t] = 1.f / (1.f + expf(-acc));
}

extern "C" void kernel_launch(void* const* d_in, const int* in_sizes, int n_in,
                              void* d_out, int out_size, void* d_ws, size_t ws_size,
                              hipStream_t stream) {
  const float* x       = (const float*)d_in[0];
  const float* W_embed = (const float*)d_in[1];
  const float* b_embed = (const float*)d_in[2];
  const float* W_diag  = (const float*)d_in[3];
  const float* W_chr   = (const float*)d_in[4];
  const float* b_chr   = (const float*)d_in[5];
  const float* W_eccp  = (const float*)d_in[6];
  const float* b_eccp  = (const float*)d_in[7];
  const float* W_ecc1  = (const float*)d_in[8];
  const float* b_ecc1  = (const float*)d_in[9];
  const float* W_ecc2  = (const float*)d_in[10];
  const float* b_ecc2  = (const float*)d_in[11];
  const float* W_coord = (const float*)d_in[12];
  const float* b_coord = (const float*)d_in[13];
  const float* W_kout  = (const float*)d_in[14];
  const float* b_kout  = (const float*)d_in[15];
  (void)in_sizes; (void)n_in; (void)out_size; (void)ws_size;

  float* out = (float*)d_out;
  float* o_emb    = out;
  float* o_metric = o_emb    + (size_t)MROWS * DE;
  float* o_chr    = o_metric + (size_t)MROWS * DD;
  float* o_scores = o_chr    + (size_t)MROWS * NC;
  float* o_idx    = o_scores + (size_t)MROWS * 32;
  float* o_minh   = o_idx    + (size_t)MROWS * 32;
  float* o_maxh   = o_minh   + (size_t)MROWS * 16;
  float* o_ecc    = o_maxh   + (size_t)MROWS * 16;
  float* o_coords = o_ecc    + (size_t)MROWS * ECCN;
  float* o_kflat  = o_coords + (size_t)MROWS * DC;

  char* ws = (char*)d_ws;
  size_t off = 0;
  auto carve = [&](size_t bytes) -> char* {
    char* p = ws + off;
    off = (off + bytes + 255) & ~(size_t)255;
    return p;
  };
  unsigned short* xbf  = (unsigned short*)carve((size_t)MROWS * DD * 2);
  unsigned short* wtE  = (unsigned short*)carve((size_t)DE * DD * 2);
  unsigned short* wtD  = (unsigned short*)carve((size_t)DD * DD * 2);
  unsigned short* wtC  = (unsigned short*)carve((size_t)NC * DD * 2);
  unsigned short* wtP  = (unsigned short*)carve((size_t)ECCN * DD * 2);
  unsigned short* wtX  = (unsigned short*)carve((size_t)DC * DD * 2);
  unsigned short* wtK  = (unsigned short*)carve((size_t)DD * KFD * 2);
  unsigned short* embn = (unsigned short*)carve((size_t)MROWS * DE * 2);
  unsigned short* kfbf = (unsigned short*)carve((size_t)MROWS * KFD * 2);
  float* pbuf = (float*)carve((size_t)MROWS * ECCN * 4);
  float* hbuf = (float*)carve((size_t)MROWS * ECCN * 2 * 4);

  // stage bf16 copies (x read once, reused by 5 GEMMs)
  {
    int n = MROWS * DD;
    k_f32_to_bf16<<<(n + 255) / 256, 256, 0, stream>>>(x, xbf, n);
  }
  k_transpose_bf16<<<(DD * DE  + 255) / 256, 256, 0, stream>>>(W_embed, wtE, DD, DE);
  k_transpose_bf16<<<(DD * DD  + 255) / 256, 256, 0, stream>>>(W_diag,  wtD, DD, DD);
  k_transpose_bf16<<<(DD * NC  + 255) / 256, 256, 0, stream>>>(W_chr,   wtC, DD, NC);
  k_transpose_bf16<<<(DD * ECCN+ 255) / 256, 256, 0, stream>>>(W_eccp,  wtP, DD, ECCN);
  k_transpose_bf16<<<(DD * DC  + 255) / 256, 256, 0, stream>>>(W_coord, wtX, DD, DC);
  k_transpose_bf16<<<(KFD * DD + 255) / 256, 256, 0, stream>>>(W_kout,  wtK, KFD, DD);

  auto gemm = [&](const unsigned short* A, const unsigned short* Bt, const float* bias,
                  float* C, int M, int N, int K) {
    int tiles = (M / 32) * (N / 32);              // one 32x32 block per wave
    k_wmma_gemm<<<(tiles + 3) / 4, 128, 0, stream>>>(A, Bt, bias, C, M, N, K);
  };
  gemm(xbf, wtE, b_embed, o_emb,    MROWS, DE,   DD);
  gemm(xbf, wtD, nullptr, o_metric, MROWS, DD,   DD);
  gemm(xbf, wtC, b_chr,   o_chr,    MROWS, NC,   DD);
  gemm(xbf, wtP, b_eccp,  pbuf,     MROWS, ECCN, DD);
  gemm(xbf, wtX, b_coord, o_coords, MROWS, DC,   DD);

  k_normalize<<<MROWS, 256, 0, stream>>>(o_emb, embn);
  k_sim_topk<<<BB * (SS / 16), 32, 0, stream>>>(embn, o_scores, o_idx, o_minh, o_maxh);

  {
    int n = MROWS * KFD;
    k_build_kflat<<<(n + 255) / 256, 256, 0, stream>>>(o_emb, o_coords, kfbf);
  }
  gemm(kfbf, wtK, b_kout, o_kflat, MROWS, DD, KFD);

  k_ecc_hidden<<<MROWS, 64, 0, stream>>>(pbuf, W_ecc1, b_ecc1, hbuf);
  k_ecc_out<<<MROWS, 32, 0, stream>>>(hbuf, W_ecc2, b_ecc2, o_ecc);
}
